// SO2_Conv1_WithRadialBlock_48782238548458
// MI455X (gfx1250) — matrix operand order repaired and checked by reading, not compiled, not run on hardware
//
#include <hip/hip_runtime.h>

// ---------------------------------------------------------------------------
// SO2 conv + radial block for MI455X (gfx1250), bf16 WMMA path.
//   GEMM0: (E x 896)  @ (896 x 1024)  + bias   -> gate(128) | out[:, 0:7]
//   GEMM1: (E x 1536) @ (1536 x 1536)          -> out[:, 7:19]   (contiguous)
//   GEMM2: (E x 1280) @ (1280 x 1280)          -> out[:, 19:29]  (contiguous)
// Weights converted once per launch to bf16, fragment-major, in d_ws.
// ---------------------------------------------------------------------------

typedef __attribute__((ext_vector_type(16))) __bf16 v16bf;
typedef __attribute__((ext_vector_type(8)))  __bf16 v8bf;
typedef __attribute__((ext_vector_type(8)))  float  v8f;

static constexpr int XROW = 29 * 128;   // 3712 floats per x row
static constexpr int EROW = 2304;       // floats per x_edge row

__device__ __forceinline__ unsigned short f2bf(float v) {
    union { __bf16 b; unsigned short u; } c;
    c.b = (__bf16)v;
    return c.u;
}

// ---------------------------------------------------------------------------
// Weight converters: emit bf16 B in fragment-major layout.
// Tile (kt, nt) occupies 32 lanes * 16 elements; lane holds cols n = nt*16+m,
// elements e give k = kt*32 + (lane>>4)*16 + e  (contiguous k order, A matches).
// ---------------------------------------------------------------------------

__global__ void conv_w0(const float* __restrict__ W, unsigned short* __restrict__ B) {
    const int KT = 896 / 32, NT = 1024 / 16;
    int t = blockIdx.x * blockDim.x + threadIdx.x;
    if (t >= KT * NT * 32) return;
    int lane = t & 31, tile = t >> 5;
    int kt = tile % KT, nt = tile / KT;
    int m = lane & 15, hf = lane >> 4;
    int n = nt * 16 + m;
    unsigned short* dst = B + ((size_t)tile * 32 + lane) * 16;
#pragma unroll
    for (int e = 0; e < 16; ++e) {
        int k = kt * 32 + hf * 16 + e;
        dst[e] = f2bf(W[(size_t)n * 896 + k]);   // B[k][n] = W[n][k]
    }
}

// SO(2) block weight: B (2S x 2S), B[k][n] = w_block[n][k],
// w_block = [[W1, -W2], [W2, W1]]  (each S x S row-major).
template <int S>
__global__ void conv_wm(const float* __restrict__ W1, const float* __restrict__ W2,
                        unsigned short* __restrict__ B) {
    const int KT = (2 * S) / 32, NT = (2 * S) / 16;
    int t = blockIdx.x * blockDim.x + threadIdx.x;
    if (t >= KT * NT * 32) return;
    int lane = t & 31, tile = t >> 5;
    int kt = tile % KT, nt = tile / KT;
    int m = lane & 15, hf = lane >> 4;
    int n = nt * 16 + m;
    unsigned short* dst = B + ((size_t)tile * 32 + lane) * 16;
#pragma unroll
    for (int e = 0; e < 16; ++e) {
        int k = kt * 32 + hf * 16 + e;
        float v;
        if (n < S) v = (k < S) ?  W1[(size_t)n * S + k]       : -W2[(size_t)n * S + (k - S)];
        else       v = (k < S) ?  W2[(size_t)(n - S) * S + k] :  W1[(size_t)(n - S) * S + (k - S)];
        dst[e] = f2bf(v);
    }
}

// ---------------------------------------------------------------------------
// Row-block-stationary bf16 WMMA GEMM.
// One workgroup (8 waves) owns 16 edges; stages bf16(x*e) full-K into LDS,
// then sweeps N in 256-col passes (2 tiles per wave), all weights from L2.
// ---------------------------------------------------------------------------
template <int K, int N, int XOFF, int EOFF, int EWRAP, int OUTOFF, bool M0MODE>
__global__ __launch_bounds__(256)
void so2_gemm(const float* __restrict__ x, const float* __restrict__ xe,
              const unsigned short* __restrict__ Bswz,
              const float* __restrict__ bias,
              float* __restrict__ out, int E) {
    constexpr int KP = K + 8;        // pad: (K+8)/2 dwords % 64 == 4 -> conflict-free
    constexpr int KT = K / 32;
    __shared__ __align__(16) unsigned short sA[16 * KP];

    const int tid = threadIdx.x;
    const int row_base = blockIdx.x * 16;

    // ---- stage A = bf16(x * e) into LDS (each edge row read from HBM once)
    for (int idx = tid; idx < 16 * (K / 2); idx += 256) {
        int r  = idx / (K / 2);
        int k2 = (idx - r * (K / 2)) * 2;
        int edge = row_base + r;
        float vx = 0.f, vy = 0.f;
        if (edge < E) {
            int ek = (EWRAP != 0 && k2 >= EWRAP) ? (k2 - EWRAP) : k2;
            float2 a = *(const float2*)(x  + (size_t)edge * XROW + XOFF + k2);
            float2 b = *(const float2*)(xe + (size_t)edge * EROW + EOFF + ek);
            vx = a.x * b.x; vy = a.y * b.y;
        }
        unsigned int pk = (unsigned int)f2bf(vx) | ((unsigned int)f2bf(vy) << 16);
        *(unsigned int*)&sA[r * KP + k2] = pk;
    }
    __syncthreads();

    const int wave = tid >> 5;
    const int lane = tid & 31;
    const int m  = lane & 15;
    const int hf = lane >> 4;
    const unsigned short* arow = &sA[m * KP + hf * 16];
    const v16bf* Bv = (const v16bf*)Bswz;

    for (int no = 0; no < N / 256; ++no) {
        int nt0 = no * 16 + wave * 2;
        v8f acc0 = {0.f,0.f,0.f,0.f,0.f,0.f,0.f,0.f};
        v8f acc1 = {0.f,0.f,0.f,0.f,0.f,0.f,0.f,0.f};
        size_t i0 = (size_t)nt0 * KT * 32 + lane;
        size_t i1 = i0 + (size_t)KT * 32;
#pragma unroll 4
        for (int kt = 0; kt < KT; ++kt) {
            // A fragment: contiguous k = hf*16 + e (matches B fragment layout)
            const v8bf* ap = (const v8bf*)(arow + kt * 32);
            v8bf lo = ap[0];
            v8bf hi = ap[1];                       // +8 bf16 elements
            v16bf a = __builtin_shufflevector(lo, hi,
                        0,1,2,3,4,5,6,7, 8,9,10,11,12,13,14,15);
            v16bf b0 = Bv[i0 + (size_t)kt * 32];
            v16bf b1 = Bv[i1 + (size_t)kt * 32];
            acc0 = __builtin_amdgcn_wmma_f32_16x16x32_bf16(
                       false, a, false, b0, (short)0, acc0, false, false);
            acc1 = __builtin_amdgcn_wmma_f32_16x16x32_bf16(
                       false, a, false, b1, (short)0, acc1, false, false);
        }
        // ---- store: D layout — lane half selects M=r or M=8+r, col = lane&15
#pragma unroll
        for (int t2 = 0; t2 < 2; ++t2) {
            const v8f& acc = t2 ? acc1 : acc0;
            int col = (nt0 + t2) * 16 + m;
            float bb = M0MODE ? bias[col] : 0.f;
            int erow0 = row_base + hf * 8;
#pragma unroll
            for (int r = 0; r < 8; ++r) {
                int erow = erow0 + r;
                if (erow < E) {
                    float val = acc[r] + bb;
                    if (M0MODE) {
                        if (col < 128)
                            out[(size_t)E * XROW + (size_t)erow * 128 + col] = val;   // gate
                        else
                            out[(size_t)erow * XROW + (col - 128)] = val;             // out[:,0:7]
                    } else {
                        out[(size_t)erow * XROW + OUTOFF + col] = val;
                    }
                }
            }
        }
    }
}

// ---------------------------------------------------------------------------
extern "C" void kernel_launch(void* const* d_in, const int* in_sizes, int n_in,
                              void* d_out, int out_size, void* d_ws, size_t ws_size,
                              hipStream_t stream) {
    const float* x    = (const float*)d_in[0];
    const float* xe   = (const float*)d_in[1];
    const float* w0   = (const float*)d_in[2];
    const float* b0   = (const float*)d_in[3];
    const float* w1_1 = (const float*)d_in[4];
    const float* w2_1 = (const float*)d_in[5];
    const float* w1_2 = (const float*)d_in[6];
    const float* w2_2 = (const float*)d_in[7];
    float* out = (float*)d_out;

    const int E = in_sizes[0] / XROW;   // 50000

    unsigned short* B0 = (unsigned short*)d_ws;                 // 896 x 1024 bf16
    unsigned short* B1 = B0 + (size_t)896 * 1024;               // 1536 x 1536 bf16
    unsigned short* B2 = B1 + (size_t)1536 * 1536;              // 1280 x 1280 bf16

    {   // weight -> bf16 fragment-major converts (weights stay L2-resident)
        int t0 = (896 / 32) * (1024 / 16) * 32;
        conv_w0<<<(t0 + 255) / 256, 256, 0, stream>>>(w0, B0);
        int t1 = (1536 / 32) * (1536 / 16) * 32;
        conv_wm<768><<<(t1 + 255) / 256, 256, 0, stream>>>(w1_1, w2_1, B1);
        int t2 = (1280 / 32) * (1280 / 16) * 32;
        conv_wm<640><<<(t2 + 255) / 256, 256, 0, stream>>>(w1_2, w2_2, B2);
    }

    const int blocks = (E + 15) / 16;   // 3125
    so2_gemm< 896, 1024,    0,    0,   0,    0, true ><<<blocks, 256, 0, stream>>>(x, xe, B0, b0,      out, E);
    so2_gemm<1536, 1536,  896,  896, 768,  896, false><<<blocks, 256, 0, stream>>>(x, xe, B1, nullptr, out, E);
    so2_gemm<1280, 1280, 2432, 1664, 640, 2432, false><<<blocks, 256, 0, stream>>>(x, xe, B2, nullptr, out, E);
}
